// Attention_11089605558362
// MI455X (gfx1250) — compile-verified, matrix operand order checked by reading, not run on hardware
//
#include <hip/hip_runtime.h>
#include <stdint.h>

// ---------------------------------------------------------------------------
// CDNA5 (gfx1250) wave32 WMMA implementation of the relu/col-norm attention.
//   B=8, L=2048, D=64.  All matmuls via v_wmma_f32_16x16x32_f16.
// Pipeline:
//   K1 proj      : BZh (row-major f16), CZh (row-major f16), VZt (transposed f16)
//   K2 zero      : denom = 0
//   K3 colsum    : denom[b,m] += sum_l relu(BZ_l . CZ_m)   (WMMA + f32 atomics)
//   K4 attn_out  : out = x + [relu(BZ.CZ^T)/(denom+eps)] . VZ
//                  CZ/VZ chunks staged to LDS via global_load_async_to_lds_b128
// ---------------------------------------------------------------------------

typedef _Float16 v16h __attribute__((ext_vector_type(16)));
typedef _Float16 h8   __attribute__((ext_vector_type(8)));
typedef float    v8f  __attribute__((ext_vector_type(8)));
typedef float    f4   __attribute__((ext_vector_type(4)));

#define WMMA_F16(a, b, c) \
  __builtin_amdgcn_wmma_f32_16x16x32_f16(false, (a), false, (b), (short)0, (c), false, false)

static constexpr int NB  = 8;     // batch
static constexpr int LL  = 2048;  // sequence
static constexpr int DD  = 64;    // head dim
static constexpr int PAD = 72;    // padded LDS row stride in halves (36 banks)
static constexpr float EPS = 1e-8f;

// --- CDNA5 async global->LDS copy (16B per lane), ISA §15.18.3 op 98 -------
__device__ __forceinline__ void async_copy_b128(uint32_t lds_byte_addr,
                                                const void* gptr) {
  asm volatile("global_load_async_to_lds_b128 %0, %1, off"
               :: "v"(lds_byte_addr),
                  "v"((unsigned long long)(uintptr_t)gptr)
               : "memory");
}
__device__ __forceinline__ void wait_async0() {
  asm volatile("s_wait_asynccnt 0" ::: "memory");
}

// --- fragment loaders -------------------------------------------------------
// A-operand (and, by transpose-symmetry, B-operand) fragment of a row-major
// f16 matrix. Per ISA 7.12.2 (16-bit A 16x32): lane = row%16, half = lane/16;
// v16h elems 0..7  = halves [k0 +    half*8 .. +7]
// v16h elems 8..15 = halves [k0 + 16+half*8 .. +7]
__device__ __forceinline__ v16h frag_h16(const _Float16* M, int ld, int row,
                                         int k0, int lane) {
  const int half = (lane >> 4) & 1;
  const _Float16* p = M + (size_t)row * ld + k0 + half * 8;
  h8 lo = *(const h8*)(p);
  h8 hi = *(const h8*)(p + 16);
  v16h f;
#pragma unroll
  for (int i = 0; i < 8; ++i) { f[i] = lo[i]; f[i + 8] = hi[i]; }
  return f;
}

// Same pattern from a row-major fp32 matrix, converting to f16 on the fly.
__device__ __forceinline__ v16h frag_f32(const float* M, int ld, int row,
                                         int k0, int lane) {
  const int half = (lane >> 4) & 1;
  const float* p = M + (size_t)row * ld + k0 + half * 8;
  f4 a0 = *(const f4*)(p);      f4 a1 = *(const f4*)(p + 4);
  f4 b0 = *(const f4*)(p + 16); f4 b1 = *(const f4*)(p + 20);
  v16h f;
#pragma unroll
  for (int i = 0; i < 4; ++i) {
    f[i]      = (_Float16)a0[i];
    f[i + 4]  = (_Float16)a1[i];
    f[i + 8]  = (_Float16)b0[i];
    f[i + 12] = (_Float16)b1[i];
  }
  return f;
}

// --- K1: projections --------------------------------------------------------
// 128 threads = 4 waves; each wave owns a 16-row strip of x (M = B*L rows).
// Y = x @ W^T + bias.  B-fragment of W^T == row fragment of W.
__global__ __launch_bounds__(128) void k1_proj(
    const float* __restrict__ x,
    const float* __restrict__ Bw, const float* __restrict__ Bb,
    const float* __restrict__ Cw, const float* __restrict__ Cb,
    const float* __restrict__ Vw, const float* __restrict__ Vb,
    _Float16* __restrict__ BZ, _Float16* __restrict__ CZ,
    _Float16* __restrict__ VZt) {
  const int lane = threadIdx.x & 31;
  const int wave = threadIdx.x >> 5;
  const int ln   = lane & 15;
  const int half = lane >> 4;
  const int rowBase = blockIdx.x * 64 + wave * 16;  // row in [0, B*L)
  const int b       = rowBase / LL;                 // whole strip in one batch

  const v16h a0 = frag_f32(x, DD, rowBase + ln, 0,  lane);
  const v16h a1 = frag_f32(x, DD, rowBase + ln, 32, lane);

  const float* Ws[3]  = {Bw, Cw, Vw};
  const float* bia[3] = {Bb, Cb, Vb};

#pragma unroll
  for (int j = 0; j < 3; ++j) {
#pragma unroll
    for (int t = 0; t < 4; ++t) {               // output-feature tile e
      const int e = t * 16 + ln;
      const v16h w0 = frag_f32(Ws[j], DD, e, 0,  lane);
      const v16h w1 = frag_f32(Ws[j], DD, e, 32, lane);
      v8f c = {};
      c = WMMA_F16(a0, w0, c);
      c = WMMA_F16(a1, w1, c);
      const float bv = bia[j][e];
      if (j < 2) {
        // D-tile: lane holds column e, element i is row m = rowBase+i+8*half.
        _Float16* dst = (j == 0) ? BZ : CZ;
#pragma unroll
        for (int i = 0; i < 8; ++i) {
          const int m = rowBase + i + 8 * half;
          dst[(size_t)m * DD + e] = (_Float16)(c[i] + bv);
        }
      } else {
        // VZ transposed: VZt[b][e][m]; element i -> consecutive m => packed store.
        h8 o;
#pragma unroll
        for (int i = 0; i < 8; ++i) o[i] = (_Float16)(c[i] + bv);
        _Float16* dst = VZt + ((size_t)b * DD + e) * LL +
                        (rowBase - b * LL) + 8 * half;
        *(h8*)dst = o;
      }
    }
  }
}

// --- K2: zero the denominators ---------------------------------------------
__global__ void k2_zero(float* __restrict__ p, int n) {
  int i = blockIdx.x * blockDim.x + threadIdx.x;
  if (i < n) p[i] = 0.0f;
}

// --- K3: column sums of relu(scores) ---------------------------------------
// grid = (L/64 row-tiles, L/64 col-tiles, B); 4 waves, one 16-row strip each.
__global__ __launch_bounds__(128) void k3_colsum(
    const _Float16* __restrict__ BZ, const _Float16* __restrict__ CZ,
    float* __restrict__ denom) {
  const int lane = threadIdx.x & 31;
  const int wave = threadIdx.x >> 5;
  const int ln   = lane & 15;
  const int b    = blockIdx.z;

  const int lRow = b * LL + blockIdx.x * 64 + wave * 16 + ln;
  const v16h a0 = frag_h16(BZ, DD, lRow, 0,  lane);
  const v16h a1 = frag_h16(BZ, DD, lRow, 32, lane);

#pragma unroll
  for (int t = 0; t < 4; ++t) {
    const int mCol = blockIdx.y * 64 + t * 16 + ln;
    const v16h c0 = frag_h16(CZ, DD, b * LL + mCol, 0,  lane);
    const v16h c1 = frag_h16(CZ, DD, b * LL + mCol, 32, lane);
    v8f s = {};
    s = WMMA_F16(a0, c0, s);
    s = WMMA_F16(a1, c1, s);
    float part = 0.0f;
#pragma unroll
    for (int i = 0; i < 8; ++i) part += fmaxf(s[i], 0.0f);
    atomicAdd(&denom[(size_t)b * LL + mCol], part);  // both halves contribute
  }
}

// --- K4: out = x + (relu(S)/denom) . VZ ------------------------------------
// grid = (L/64, B); 4 waves, each a 16-row l-strip; stream m in chunks of 64.
// CZ/VZt chunks (8 KB each) staged once per WG via async DMA to LDS; the
// normalized score tile bounces registers -> LDS -> f16 A-fragment.
__global__ __launch_bounds__(128) void k4_attn_out(
    const float* __restrict__ x,
    const _Float16* __restrict__ BZ, const _Float16* __restrict__ CZ,
    const _Float16* __restrict__ VZt, const float* __restrict__ denom,
    float* __restrict__ out) {
  __shared__ _Float16 sCZ[64 * PAD];   // staged CZ chunk  (rows m, ld=72)
  __shared__ _Float16 sVZ[64 * PAD];   // staged VZt chunk (rows d, ld=72)
  __shared__ _Float16 sS[4][16 * PAD]; // per-wave 16x64 normalized score tile

  const int tid  = threadIdx.x;
  const int lane = tid & 31;
  const int wave = tid >> 5;
  const int ln   = lane & 15;
  const int half = lane >> 4;
  const int b    = blockIdx.y;
  const int lTile = blockIdx.x * 64 + wave * 16;   // l within batch
  const int lRow  = b * LL + lTile + ln;

  const v16h a0 = frag_h16(BZ, DD, lRow, 0,  lane);
  const v16h a1 = frag_h16(BZ, DD, lRow, 32, lane);

  const _Float16* vzb = VZt + (size_t)b * DD * LL;
  _Float16* myS = &sS[wave][0];

  v8f acc[4] = {};  // 16 x 64 output accumulator (4 d-tiles)

  for (int mc = 0; mc < LL; mc += 64) {
    // ---- stage CZ[mc..mc+63] and VZt[:, mc..mc+63] via async DMA ----
    // 512 x 16B chunks per matrix; 128 threads x 4 each.
#pragma unroll
    for (int c = 0; c < 4; ++c) {
      const int ck  = tid * 4 + c;       // 0..511
      const int row = ck >> 3;           // 0..63
      const int sub = ck & 7;            // 16B sub-chunk within the 128B row
      async_copy_b128((uint32_t)(uintptr_t)&sCZ[row * PAD + sub * 8],
                      CZ + ((size_t)(b * LL + mc + row)) * DD + sub * 8);
      async_copy_b128((uint32_t)(uintptr_t)&sVZ[row * PAD + sub * 8],
                      vzb + (size_t)row * LL + mc + sub * 8);
    }
    wait_async0();        // this wave's copies landed in LDS
    __syncthreads();      // all waves' copies visible

    // ---- score tiles from LDS: relu, column-scale, spill to sS ----
#pragma unroll
    for (int t = 0; t < 4; ++t) {
      const v16h c0 = frag_h16(sCZ, PAD, t * 16 + ln, 0,  lane);
      const v16h c1 = frag_h16(sCZ, PAD, t * 16 + ln, 32, lane);
      v8f s = {};
      s = WMMA_F16(a0, c0, s);
      s = WMMA_F16(a1, c1, s);
      const float r = 1.0f / (denom[(size_t)b * LL + mc + t * 16 + ln] + EPS);
#pragma unroll
      for (int i = 0; i < 8; ++i) {
        const float v = fmaxf(s[i], 0.0f) * r;
        myS[(i + 8 * half) * PAD + t * 16 + ln] = (_Float16)v;  // row-major (l,m)
      }
    }
    __syncthreads();

    // ---- O += S_f16 . VZ : A-frags from sS, B-frags from sVZ ----
    const v16h sa0 = frag_h16(myS, PAD, ln, 0,  lane);
    const v16h sa1 = frag_h16(myS, PAD, ln, 32, lane);
#pragma unroll
    for (int t = 0; t < 4; ++t) {
      const v16h vb0 = frag_h16(sVZ, PAD, t * 16 + ln, 0,  lane);
      const v16h vb1 = frag_h16(sVZ, PAD, t * 16 + ln, 32, lane);
      acc[t] = WMMA_F16(sa0, vb0, acc[t]);
      acc[t] = WMMA_F16(sa1, vb1, acc[t]);
    }
    __syncthreads();      // done reading staged chunks; safe to overwrite
  }

  // ---- epilogue: residual add, D-tile scatter ----
#pragma unroll
  for (int t = 0; t < 4; ++t) {
#pragma unroll
    for (int i = 0; i < 8; ++i) {
      const int l = lTile + i + 8 * half;
      const int d = t * 16 + ln;
      const size_t idx = ((size_t)b * LL + l) * DD + d;
      out[idx] = x[idx] + acc[t][i];
    }
  }
}

// ---------------------------------------------------------------------------
extern "C" void kernel_launch(void* const* d_in, const int* in_sizes, int n_in,
                              void* d_out, int out_size, void* d_ws, size_t ws_size,
                              hipStream_t stream) {
  (void)in_sizes; (void)n_in; (void)out_size; (void)ws_size;
  const float* x  = (const float*)d_in[0];
  const float* Bw = (const float*)d_in[1];
  const float* Bb = (const float*)d_in[2];
  const float* Cw = (const float*)d_in[3];
  const float* Cb = (const float*)d_in[4];
  const float* Vw = (const float*)d_in[5];
  const float* Vb = (const float*)d_in[6];
  float* out = (float*)d_out;

  const size_t M = (size_t)NB * LL;          // 16384 rows
  _Float16* BZ  = (_Float16*)d_ws;           // M x 64 f16 (2 MB)
  _Float16* CZ  = BZ + M * DD;               // M x 64 f16 (2 MB)
  _Float16* VZt = CZ + M * DD;               // B x 64 x L f16, transposed (2 MB)
  float*    den = (float*)(VZt + M * DD);    // B x L f32 (64 KB)

  k1_proj<<<dim3(M / 64), 128, 0, stream>>>(x, Bw, Bb, Cw, Cb, Vw, Vb, BZ, CZ, VZt);
  k2_zero<<<dim3((NB * LL + 255) / 256), 256, 0, stream>>>(den, NB * LL);
  k3_colsum<<<dim3(LL / 64, LL / 64, NB), 128, 0, stream>>>(BZ, CZ, den);
  k4_attn_out<<<dim3(LL / 64, NB), 128, 0, stream>>>(x, BZ, CZ, VZt, den, out);
}
